// Seq2SeqModel_23398981828984
// MI455X (gfx1250) — matrix operand order repaired and checked by reading, not compiled, not run on hardware
//
#include <hip/hip_runtime.h>
#include <stdint.h>

// ---------------------------------------------------------------------------
// CDNA5 (gfx1250) seq2seq: bf16 WMMA for all GEMMs, f32 accumulate.
// T=32, B=32, H=512, Hd=256, L=2, V=32000, BEAM=100.
// Round 2: GEMM staging now uses GLOBAL_LOAD_ASYNC_TO_LDS_B128 (ASYNCcnt)
// instead of global_load_b128 + ds_store_b128 (no VGPR bounce).
// ---------------------------------------------------------------------------

typedef __attribute__((ext_vector_type(16))) __bf16 v16bf;
typedef __attribute__((ext_vector_type(8)))  float  v8f;

struct U32x8 { unsigned int d[8]; };

__device__ inline unsigned short f2bf(float f) {
  unsigned int u = __float_as_uint(f);
  u = u + 0x7FFFu + ((u >> 16) & 1u);          // round-to-nearest-even
  return (unsigned short)(u >> 16);
}
__device__ inline float sigm(float x) { return 1.0f / (1.0f + expf(-x)); }

// Async direct global->LDS copy of 16 bytes per lane (CDNA5 TDM-lite path).
// INST_OFFSET is added to both the LDS and the global address (ISA 15.18.3).
__device__ inline void async_g2l_b128(unsigned lds_addr, unsigned long long gaddr) {
  asm volatile("global_load_async_to_lds_b128 %0, %1, off"
               :: "v"(lds_addr), "v"(gaddr) : "memory");
}
__device__ inline void async_g2l_b128_off16(unsigned lds_addr, unsigned long long gaddr) {
  asm volatile("global_load_async_to_lds_b128 %0, %1, off offset:16"
               :: "v"(lds_addr), "v"(gaddr) : "memory");
}
__device__ inline void wait_async0() {
  asm volatile("s_wait_asynccnt 0x0" ::: "memory");
}

// A-fragment (16x32 bf16, MxK) from row-major ushort array. base = &A[tileM0][k0].
// Layout (ISA 7.12.2): lanes 0-15: VGPR v -> K = {2v,2v+1} (v<4) or {16+2(v-4),...};
// lanes 16-31 add +8 to K. As dwords: dw = v + (v>=4?4:0) + (lane>=16?4:0).
__device__ inline v16bf load_a_frag(const unsigned short* base, int ld) {
  const int lane = threadIdx.x & 31;
  const int hh = lane >> 4;
  const unsigned int* rp = (const unsigned int*)(base + (size_t)(lane & 15) * ld);
  U32x8 u;
#pragma unroll
  for (int v = 0; v < 8; ++v) u.d[v] = rp[v + ((v >= 4) ? 4 : 0) + hh * 4];
  return __builtin_bit_cast(v16bf, u);
}
// B-fragment (32x16 bf16, KxN): lane supplies column N=lane&15, K in A's table order. LDS version.
__device__ inline v16bf load_b_frag_lds(const unsigned short* base, int ld) {
  const int lane = threadIdx.x & 31;
  const int hh = lane >> 4;
  const unsigned short* p = base + (lane & 15);
  U32x8 u;
#pragma unroll
  for (int v = 0; v < 8; ++v) {
    int k0 = 2 * v + ((v >= 4) ? 8 : 0) + hh * 8;
    u.d[v] = (unsigned int)p[(size_t)k0 * ld] | ((unsigned int)p[(size_t)(k0 + 1) * ld] << 16);
  }
  return __builtin_bit_cast(v16bf, u);
}
// B-fragment straight from global (weights pre-transposed to [K][N] bf16): column N=lane&15, K in table order.
__device__ inline v16bf load_b_frag_g(const unsigned short* base, int ld) {
  const int lane = threadIdx.x & 31;
  const int hh = lane >> 4;
  const unsigned short* p = base + (lane & 15);
  U32x8 u;
#pragma unroll
  for (int v = 0; v < 8; ++v) {
    int k0 = 2 * v + ((v >= 4) ? 8 : 0) + hh * 8;
    u.d[v] = (unsigned int)p[(size_t)k0 * ld] | ((unsigned int)p[(size_t)(k0 + 1) * ld] << 16);
  }
  return __builtin_bit_cast(v16bf, u);
}
__device__ inline v8f wmma_bf16(v16bf a, v16bf b, v8f c) {
  return __builtin_amdgcn_wmma_f32_16x16x32_bf16(false, a, false, b, (short)0, c,
                                                 false, false);
}

// ---------------------------------------------------------------------------
// Weight transpose + f32->bf16:  D[k*ld + n] = bf16(W[n*K + k])
// ---------------------------------------------------------------------------
__global__ void tconv_kernel(const float* __restrict__ W, unsigned short* __restrict__ D,
                             int N, int K, int ld) {
  int i = blockIdx.x * 256 + threadIdx.x;
  if (i >= N * K) return;
  int n = i / K, k = i - n * K;
  D[(size_t)k * ld + n] = f2bf(W[i]);
}

__global__ void f2bf_kernel(const float* __restrict__ s, unsigned short* __restrict__ d,
                            int n) {
  int i = blockIdx.x * 256 + threadIdx.x;
  if (i < n) d[i] = f2bf(s[i]);
}

__global__ void embed_kernel(const int* __restrict__ tok, const float* __restrict__ emb,
                             float* __restrict__ x) {
  int tb = blockIdx.x;                      // t*32 + b
  int token = tok[tb];
  const float* src = emb + (size_t)token * 512;
  float* dst = x + (size_t)tb * 512;
  for (int h = threadIdx.x; h < 512; h += blockDim.x) dst[h] = src[h];
}

// ---------------------------------------------------------------------------
// Persistent single-WGP LSTM layer. 512 threads = 16 waves.
// h state lives in LDS (f32); c state lives in registers (static task map).
// Per timestep: stage [x_t ; h] as bf16 in LDS, then WMMA the 4 gate GEMMs.
// ---------------------------------------------------------------------------
template <int HH>
__global__ void __launch_bounds__(512)
lstm_layer_kernel(const float* __restrict__ x_in, const unsigned short* __restrict__ Wt,
                  const float* __restrict__ bias, const float* __restrict__ h0,
                  const float* __restrict__ c0, float* __restrict__ y_out, int colOff,
                  float* __restrict__ hT, float* __restrict__ cT, int reverse) {
  constexpr int Kin = 512, K = Kin + HH, G = 4 * HH;
  constexpr int TPW = (2 * (HH / 16)) / 16;       // tasks per wave (2 or 4)
  extern __shared__ unsigned char smem[];
  float* hS = (float*)smem;                       // [32*HH] f32
  unsigned short* xh = (unsigned short*)(smem + (size_t)32 * HH * 4);  // [32*K] bf16
  const int tid = threadIdx.x, lane = tid & 31, wave = tid >> 5;
  const int half8 = (lane >> 4) << 3, nidx = lane & 15;

  for (int i = tid; i < 32 * HH; i += 512)
    hS[i] = h0 ? h0[(i / HH) * 512 + (i % HH)] : 0.0f;

  float creg[TPW][8];
#pragma unroll
  for (int ti = 0; ti < TPW; ++ti) {
    int task = wave + ti * 16;
    int mtile = task & 1, jbase = (task >> 1) * 16;
    int bb0 = mtile * 16 + half8, j = jbase + nidx;
#pragma unroll
    for (int e = 0; e < 8; ++e)
      creg[ti][e] = c0 ? c0[(size_t)(bb0 + e) * 512 + j] : 0.0f;
  }
  __syncthreads();

  for (int t = 0; t < 32; ++t) {
    const int xt = reverse ? (31 - t) : t;
    for (int i = tid; i < 32 * K; i += 512) {
      int row = i / K, col = i - row * K;
      float v = (col < Kin) ? x_in[((size_t)xt * 32 + row) * 512 + col]
                            : hS[row * HH + (col - Kin)];
      xh[i] = f2bf(v);
    }
    __syncthreads();
#pragma unroll
    for (int ti = 0; ti < TPW; ++ti) {
      int task = wave + ti * 16;
      int mtile = task & 1, jbase = (task >> 1) * 16;
      v8f zz = {0, 0, 0, 0, 0, 0, 0, 0};
      v8f ai = zz, af = zz, ag = zz, ao = zz;
      for (int kk = 0; kk < K; kk += 32) {
        v16bf a = load_a_frag(xh + (size_t)(mtile * 16) * K + kk, K);
        const unsigned short* wb = Wt + (size_t)kk * G + jbase;
        v16bf b0 = load_b_frag_g(wb, G);
        v16bf b1 = load_b_frag_g(wb + HH, G);
        v16bf b2 = load_b_frag_g(wb + 2 * HH, G);
        v16bf b3 = load_b_frag_g(wb + 3 * HH, G);
        ai = wmma_bf16(a, b0, ai);
        af = wmma_bf16(a, b1, af);
        ag = wmma_bf16(a, b2, ag);
        ao = wmma_bf16(a, b3, ao);
      }
      int j = jbase + nidx, bb0 = mtile * 16 + half8;
      float bi_ = bias[0 * HH + j], bf_ = bias[1 * HH + j];
      float bg_ = bias[2 * HH + j], bo_ = bias[3 * HH + j];
#pragma unroll
      for (int e = 0; e < 8; ++e) {
        float gi = ai[e] + bi_, gf = af[e] + bf_, gg = ag[e] + bg_, go = ao[e] + bo_;
        float cn = sigm(gf) * creg[ti][e] + sigm(gi) * tanhf(gg);
        float hn = sigm(go) * tanhf(cn);
        creg[ti][e] = cn;
        int bb = bb0 + e;
        hS[bb * HH + j] = hn;
        y_out[((size_t)xt * 32 + bb) * 512 + colOff + j] = hn;
      }
    }
    __syncthreads();
  }
  if (hT)
    for (int i = tid; i < 32 * HH; i += 512) hT[(i / HH) * 512 + (i % HH)] = hS[i];
  if (cT) {
#pragma unroll
    for (int ti = 0; ti < TPW; ++ti) {
      int task = wave + ti * 16;
      int mtile = task & 1, jbase = (task >> 1) * 16;
      int j = jbase + nidx, bb0 = mtile * 16 + half8;
#pragma unroll
      for (int e = 0; e < 8; ++e) cT[(size_t)(bb0 + e) * 512 + j] = creg[ti][e];
    }
  }
}

// ---------------------------------------------------------------------------
// LDS-tiled bf16 GEMM: C[M,N] = A[M,K] @ Bt[K,N] (+bias) (opt tanh).
// Block tile 64x128, BK=32. 8 waves as 2x4; wave tile 32x32 (4 WMMA accums).
// Staging: GLOBAL_LOAD_ASYNC_TO_LDS_B128 (no VGPR bounce), ASYNCcnt-gated.
// M%64==0, N%128==0, K%32==0 for all call sites.
// ---------------------------------------------------------------------------
__global__ void __launch_bounds__(256)
gemm_bf16_kernel(const unsigned short* __restrict__ A, const unsigned short* __restrict__ Bt,
                 const float* __restrict__ bias, float* __restrict__ C,
                 int M, int N, int K, int act) {
  __shared__ __align__(16) unsigned short As[64 * 32];
  __shared__ __align__(16) unsigned short Bs[32 * 128];
  const int tid = threadIdx.x, lane = tid & 31, wave = tid >> 5;
  const int m0 = blockIdx.y * 64, n0 = blockIdx.x * 128;
  const int wm = (wave >> 2) * 32, wn = (wave & 3) * 32;
  const int half8 = (lane >> 4) << 3, nidx = lane & 15;
  v8f zz = {0, 0, 0, 0, 0, 0, 0, 0};
  v8f acc[2][2];
  acc[0][0] = zz; acc[0][1] = zz; acc[1][0] = zz; acc[1][1] = zz;

  const int arow = tid >> 2, aseg = tid & 3;   // A: 64 rows x 4 segs of 8 halfs
  const int brow = tid >> 3, bseg = tid & 7;   // B: 32 rows x 8 segs of 16 halfs
  const unsigned aLds = (unsigned)(uintptr_t)(As + arow * 32 + aseg * 8);
  const unsigned bLds = (unsigned)(uintptr_t)(Bs + brow * 128 + bseg * 16);

  for (int k0 = 0; k0 < K; k0 += 32) {
    async_g2l_b128(aLds,
        (unsigned long long)(A + (size_t)(m0 + arow) * K + k0 + aseg * 8));
    const unsigned short* bsrc = Bt + (size_t)(k0 + brow) * N + n0 + bseg * 16;
    async_g2l_b128(bLds, (unsigned long long)bsrc);
    async_g2l_b128_off16(bLds, (unsigned long long)bsrc);
    if (k0 + 32 < K)  // warm L2 for the next K-slab of B
      __builtin_prefetch(Bt + (size_t)(k0 + 32 + brow) * N + n0 + bseg * 16, 0, 1);
    wait_async0();
    __syncthreads();
    v16bf a0 = load_a_frag(As + (wm + 0) * 32, 32);
    v16bf a1 = load_a_frag(As + (wm + 16) * 32, 32);
    v16bf b0 = load_b_frag_lds(Bs + wn, 128);
    v16bf b1 = load_b_frag_lds(Bs + wn + 16, 128);
    acc[0][0] = wmma_bf16(a0, b0, acc[0][0]);
    acc[0][1] = wmma_bf16(a0, b1, acc[0][1]);
    acc[1][0] = wmma_bf16(a1, b0, acc[1][0]);
    acc[1][1] = wmma_bf16(a1, b1, acc[1][1]);
    __syncthreads();
  }
#pragma unroll
  for (int mt = 0; mt < 2; ++mt)
#pragma unroll
    for (int nt = 0; nt < 2; ++nt) {
      int n = n0 + wn + nt * 16 + nidx;
      float bv = bias ? bias[n] : 0.0f;
#pragma unroll
      for (int e = 0; e < 8; ++e) {
        int m = m0 + wm + mt * 16 + half8 + e;
        float v = acc[mt][nt][e] + bv;
        if (act) v = tanhf(v);
        C[(size_t)m * N + n] = v;
      }
    }
}

// ---------------------------------------------------------------------------
// Attention: scores, masked softmax over s, context.
// ---------------------------------------------------------------------------
__global__ void attn_scores_kernel(const float* __restrict__ xd, const float* __restrict__ mem,
                                   float* __restrict__ sc) {
  int b = blockIdx.x & 31, t = blockIdx.x >> 5, s = threadIdx.x;  // 32 threads
  const float* xr = xd + ((size_t)t * 32 + b) * 512;
  const float* mr = mem + ((size_t)s * 32 + b) * 512;
  float acc = 0.f;
  for (int h = 0; h < 512; ++h) acc += xr[h] * mr[h];
  sc[((size_t)b * 32 + t) * 32 + s] = acc;
}

__global__ void attn_softmax_kernel(float* __restrict__ sc, const int* __restrict__ len) {
  int i = blockIdx.x * blockDim.x + threadIdx.x;
  if (i >= 1024) return;                 // i = b*32 + t
  int b = i >> 5;
  float* row = sc + (size_t)i * 32;
  int ml = len[b]; ml = ml < 1 ? 1 : (ml > 32 ? 32 : ml);
  float mx = -1e30f;
  for (int s = 0; s < 32; ++s) { float v = s < ml ? row[s] : -1e9f; mx = v > mx ? v : mx; }
  float e[32], sum = 0.f;
  for (int s = 0; s < 32; ++s) { float v = s < ml ? row[s] : -1e9f; e[s] = expf(v - mx); sum += e[s]; }
  float inv = 1.0f / sum;
  for (int s = 0; s < 32; ++s) row[s] = e[s] * inv;
}

__global__ void attn_ctx_kernel(const float* __restrict__ alpha, const float* __restrict__ mem,
                                float* __restrict__ ctx) {
  int i = blockIdx.x * blockDim.x + threadIdx.x;
  if (i >= 32 * 32 * 512) return;
  int m = i >> 9, h = i & 511;
  int t = m >> 5, b = m & 31;
  const float* ar = alpha + ((size_t)b * 32 + t) * 32;
  float acc = 0.f;
  for (int s = 0; s < 32; ++s) acc += ar[s] * mem[((size_t)s * 32 + b) * 512 + h];
  ctx[i] = acc;
}

__global__ void build_cx_kernel(const float* __restrict__ ctx, const float* __restrict__ xd,
                                unsigned short* __restrict__ cx) {
  int i = blockIdx.x * blockDim.x + threadIdx.x;
  if (i >= 1024 * 1024) return;
  int m = i >> 10, d = i & 1023;
  float v = (d < 512) ? ctx[(size_t)m * 512 + d] : xd[(size_t)m * 512 + (d - 512)];
  cx[i] = f2bf(v);
}

// ---------------------------------------------------------------------------
// In-place log-softmax over V=32000; one block per (t,b) row.
// ---------------------------------------------------------------------------
__global__ void __launch_bounds__(256) log_softmax_kernel(float* __restrict__ out) {
  __shared__ float red[256];
  float* row = out + (size_t)blockIdx.x * 32000;
  int tid = threadIdx.x;
  float mx = -1e30f;
  for (int v = tid; v < 32000; v += 256) { float x = row[v]; mx = x > mx ? x : mx; }
  red[tid] = mx; __syncthreads();
  for (int s = 128; s > 0; s >>= 1) {
    if (tid < s && red[tid + s] > red[tid]) red[tid] = red[tid + s];
    __syncthreads();
  }
  mx = red[0]; __syncthreads();
  float sum = 0.f;
  for (int v = tid; v < 32000; v += 256) sum += expf(row[v] - mx);
  red[tid] = sum; __syncthreads();
  for (int s = 128; s > 0; s >>= 1) {
    if (tid < s) red[tid] += red[tid + s];
    __syncthreads();
  }
  float lse = mx + logf(red[0]);
  __syncthreads();
  for (int v = tid; v < 32000; v += 256) row[v] = row[v] - lse;
}

// ---------------------------------------------------------------------------
// Beam search, one block per batch element. Exact reduction: because
// cand[i,v] = scores[i]*nll[v] with scores>=0, the global top-100 lives in
// the per-step top-100 of nll[t]; select 100 from the 100x100 product table.
// Ties broken by ascending flat index (i*V + v), matching lax.top_k.
// ---------------------------------------------------------------------------
__global__ void __launch_bounds__(256)
beam_kernel(const float* __restrict__ logp, float* __restrict__ tok_out) {
  const int b = blockIdx.x, tid = threadIdx.x;
  __shared__ float vVal[100]; __shared__ int vIdx[100];
  __shared__ float scores[100]; __shared__ float nScores[100];
  __shared__ int parent[100]; __shared__ int ntok[100];
  __shared__ int seqs[2][100][32];
  __shared__ float rv[256]; __shared__ int ri[256];
  __shared__ float lastV; __shared__ int lastI;
  int cur = 0;
  for (int t = 0; t < 32; ++t) {
    const float* lp = logp + ((size_t)t * 32 + b) * 32000;
    if (tid == 0) { lastV = -1e30f; lastI = -1; }
    __syncthreads();
    for (int r = 0; r < 100; ++r) {            // 100 smallest of nll[t]
      float LV = lastV; int LI = lastI;
      float bv = 3.4e38f; int bi = 0x7fffffff;
      for (int v = tid; v < 32000; v += 256) {
        float val = -lp[v];
        if ((val > LV) || (val == LV && v > LI))
          if ((val < bv) || (val == bv && v < bi)) { bv = val; bi = v; }
      }
      rv[tid] = bv; ri[tid] = bi; __syncthreads();
      for (int s = 128; s > 0; s >>= 1) {
        if (tid < s && ((rv[tid + s] < rv[tid]) ||
                        (rv[tid + s] == rv[tid] && ri[tid + s] < ri[tid]))) {
          rv[tid] = rv[tid + s]; ri[tid] = ri[tid + s];
        }
        __syncthreads();
      }
      if (tid == 0) { vVal[r] = rv[0]; vIdx[r] = ri[0]; lastV = rv[0]; lastI = ri[0]; }
      __syncthreads();
    }
    if (t == 0) {
      for (int i = tid; i < 100 * 32; i += 256) seqs[0][i >> 5][i & 31] = 0;
      __syncthreads();
      if (tid < 100) { scores[tid] = vVal[tid]; seqs[0][tid][0] = vIdx[tid]; }
      cur = 0;
      __syncthreads();
    } else {
      if (tid == 0) { lastV = -1e30f; lastI = -1; }
      __syncthreads();
      for (int r = 0; r < 100; ++r) {          // 100 smallest of 100x100 products
        float LV = lastV; int LI = lastI;
        float bv = 3.4e38f; int bi = 0x7fffffff;
        for (int c = tid; c < 10000; c += 256) {
          int i = c / 100, jj = c - 100 * i;
          float val = scores[i] * vVal[jj];
          int fi = i * 32000 + vIdx[jj];
          if ((val > LV) || (val == LV && fi > LI))
            if ((val < bv) || (val == bv && fi < bi)) { bv = val; bi = fi; }
        }
        rv[tid] = bv; ri[tid] = bi; __syncthreads();
        for (int s = 128; s > 0; s >>= 1) {
          if (tid < s && ((rv[tid + s] < rv[tid]) ||
                          (rv[tid + s] == rv[tid] && ri[tid + s] < ri[tid]))) {
            rv[tid] = rv[tid + s]; ri[tid] = ri[tid + s];
          }
          __syncthreads();
        }
        if (tid == 0) {
          nScores[r] = rv[0]; parent[r] = ri[0] / 32000; ntok[r] = ri[0] % 32000;
          lastV = rv[0]; lastI = ri[0];
        }
        __syncthreads();
      }
      int nxt = cur ^ 1;
      for (int i = tid; i < 100 * 32; i += 256) {
        int r = i >> 5, p = i & 31;
        seqs[nxt][r][p] = seqs[cur][parent[r]][p];
      }
      __syncthreads();
      if (tid < 100) { seqs[nxt][tid][t] = ntok[tid]; scores[tid] = nScores[tid]; }
      cur = nxt;
      __syncthreads();
    }
  }
  for (int t = tid; t < 32; t += 256) tok_out[(size_t)t * 32 + b] = (float)seqs[cur][0][t];
}

// ---------------------------------------------------------------------------
extern "C" void kernel_launch(void* const* d_in, const int* in_sizes, int n_in,
                              void* d_out, int out_size, void* d_ws, size_t ws_size,
                              hipStream_t stream) {
  (void)in_sizes; (void)n_in; (void)out_size; (void)ws_size;
  const int* enc_src = (const int*)d_in[0];
  const int* dec_tgt = (const int*)d_in[2];
  const int* tgt_len = (const int*)d_in[3];
  const float* enc_emb = (const float*)d_in[5];
  const float* dec_emb = (const float*)d_in[6];
  const float* enc_Wih = (const float*)d_in[7];
  const float* enc_Whh = (const float*)d_in[8];
  const float* enc_b = (const float*)d_in[9];
  const float* dec_Wih = (const float*)d_in[10];
  const float* dec_Whh = (const float*)d_in[11];
  const float* dec_b = (const float*)d_in[12];
  const float* attn_Wc = (const float*)d_in[13];
  const float* out_W = (const float*)d_in[14];
  const float* out_b = (const float*)d_in[15];
  float* out = (float*)d_out;

  unsigned char* wsb = (unsigned char*)d_ws;
  size_t off = 0;
  auto alloc = [&](size_t bytes) -> void* {
    off = (off + 255) & ~(size_t)255;
    void* p = wsb + off; off += bytes; return p;
  };
  float* encA = (float*)alloc(2097152);            // [T,B,H]
  float* encB = (float*)alloc(2097152);
  float* decA = (float*)alloc(2097152);
  float* decB = (float*)alloc(2097152);
  float* hs   = (float*)alloc(131072);             // [L,B,H]
  float* cs   = (float*)alloc(131072);
  float* alpha = (float*)alloc(131072);            // [B,T,T]
  float* ctx  = (float*)alloc(2097152);
  float* attnh = (float*)alloc(2097152);
  unsigned short* cxb = (unsigned short*)alloc(2097152);       // [1024,1024] bf16
  unsigned short* attnhb = (unsigned short*)alloc(1048576);    // [1024,512] bf16
  unsigned short* encWt = (unsigned short*)alloc(6291456);     // 4 x [768,1024] bf16
  unsigned short* decWt = (unsigned short*)alloc(8388608);     // 2 x [1024,2048] bf16
  unsigned short* awct = (unsigned short*)alloc(1048576);      // [1024,512] bf16
  unsigned short* outWt = (unsigned short*)alloc(32768000);    // [512,32000] bf16

  const size_t ls256 = 32 * 256 * 4 + 32 * 768 * 2;   // 80 KB
  const size_t ls512 = 32 * 512 * 4 + 32 * 1024 * 2;  // 128 KB
  (void)hipFuncSetAttribute(reinterpret_cast<const void*>(&lstm_layer_kernel<256>),
                            hipFuncAttributeMaxDynamicSharedMemorySize, (int)ls256);
  (void)hipFuncSetAttribute(reinterpret_cast<const void*>(&lstm_layer_kernel<512>),
                            hipFuncAttributeMaxDynamicSharedMemorySize, (int)ls512);

  auto tc = [&](const float* W, unsigned short* D, int N, int K, int ld) {
    int total = N * K;
    tconv_kernel<<<(total + 255) / 256, 256, 0, stream>>>(W, D, N, K, ld);
  };
  // Encoder weights: combined [Kin+Hd, 4Hd] transposed bf16 per (layer,dir).
  for (int l = 0; l < 2; ++l)
    for (int d = 0; d < 2; ++d) {
      unsigned short* dst = encWt + (size_t)(l * 2 + d) * 768 * 1024;
      tc(enc_Wih + (size_t)(l * 2 + d) * 1024 * 512, dst, 1024, 512, 1024);
      tc(enc_Whh + (size_t)(l * 2 + d) * 1024 * 256, dst + (size_t)512 * 1024, 1024, 256, 1024);
    }
  for (int l = 0; l < 2; ++l) {
    unsigned short* dst = decWt + (size_t)l * 1024 * 2048;
    tc(dec_Wih + (size_t)l * 2048 * 512, dst, 2048, 512, 2048);
    tc(dec_Whh + (size_t)l * 2048 * 512, dst + (size_t)512 * 2048, 2048, 512, 2048);
  }
  tc(attn_Wc, awct, 512, 1024, 512);
  tc(out_W, outWt, 32000, 512, 32000);

  embed_kernel<<<1024, 128, 0, stream>>>(enc_src, enc_emb, encA);
  embed_kernel<<<1024, 128, 0, stream>>>(dec_tgt, dec_emb, decA);

  // Encoder: layer0 fwd/bwd concat into encB, layer1 concat into encA (mem bank).
  lstm_layer_kernel<256><<<1, 512, ls256, stream>>>(
      encA, encWt + (size_t)0 * 768 * 1024, enc_b + 0 * 1024, nullptr, nullptr,
      encB, 0, hs + 0 * 32 * 512 + 0, cs + 0 * 32 * 512 + 0, 0);
  lstm_layer_kernel<256><<<1, 512, ls256, stream>>>(
      encA, encWt + (size_t)1 * 768 * 1024, enc_b + 1 * 1024, nullptr, nullptr,
      encB, 256, hs + 0 * 32 * 512 + 256, cs + 0 * 32 * 512 + 256, 1);
  lstm_layer_kernel<256><<<1, 512, ls256, stream>>>(
      encB, encWt + (size_t)2 * 768 * 1024, enc_b + 2 * 1024, nullptr, nullptr,
      encA, 0, hs + 1 * 32 * 512 + 0, cs + 1 * 32 * 512 + 0, 0);
  lstm_layer_kernel<256><<<1, 512, ls256, stream>>>(
      encB, encWt + (size_t)3 * 768 * 1024, enc_b + 3 * 1024, nullptr, nullptr,
      encA, 256, hs + 1 * 32 * 512 + 256, cs + 1 * 32 * 512 + 256, 1);

  // Decoder: layer0 decA->decB, layer1 decB->decA; initial states from encoder.
  lstm_layer_kernel<512><<<1, 512, ls512, stream>>>(
      decA, decWt, dec_b, hs + 0 * 32 * 512, cs + 0 * 32 * 512,
      decB, 0, nullptr, nullptr, 0);
  lstm_layer_kernel<512><<<1, 512, ls512, stream>>>(
      decB, decWt + (size_t)1024 * 2048, dec_b + 2048, hs + 1 * 32 * 512,
      cs + 1 * 32 * 512, decA, 0, nullptr, nullptr, 0);

  // Attention + combine (tanh(Wc [ctx;x])) via WMMA GEMM.
  attn_scores_kernel<<<1024, 32, 0, stream>>>(decA, encA, alpha);
  attn_softmax_kernel<<<4, 256, 0, stream>>>(alpha, tgt_len);
  attn_ctx_kernel<<<2048, 256, 0, stream>>>(alpha, encA, ctx);
  build_cx_kernel<<<4096, 256, 0, stream>>>(ctx, decA, cxb);
  gemm_bf16_kernel<<<dim3(4, 16), 256, 0, stream>>>(cxb, awct, nullptr, attnh,
                                                    1024, 512, 1024, 1);
  f2bf_kernel<<<2048, 256, 0, stream>>>(attnh, attnhb, 524288);

  // Vocab projection (33.5 GFLOP — the roofline hot spot) into d_out logits.
  gemm_bf16_kernel<<<dim3(250, 16), 256, 0, stream>>>(attnhb, outWt, out_b, out,
                                                      1024, 32000, 512, 0);
  log_softmax_kernel<<<1024, 256, 0, stream>>>(out);
  beam_kernel<<<32, 256, 0, stream>>>(out, out + (size_t)1024 * 32000);
}
